// STAttnLayer_10763188044506
// MI455X (gfx1250) — compile-verified, matrix operand order checked by reading, not run on hardware
//
#include <hip/hip_runtime.h>

#define DEVINL __device__ __forceinline__

typedef __attribute__((ext_vector_type(16))) _Float16 v16h;
typedef __attribute__((ext_vector_type(8)))  _Float16 h8;
typedef __attribute__((ext_vector_type(8)))  float    v8f;
typedef __attribute__((ext_vector_type(8)))  float    f8v;

constexpr int   Bb = 4, Ss = 512, Jj = 24, Hh = 128, Nn = 8, Dd = 16, Ffn = 256;
constexpr float SCALE = 0.25f;   // 1/sqrt(HS=16)
constexpr float NEGF  = -1e30f;

DEVINL int lane() { return (int)(threadIdx.x & 31u); }

DEVINL v8f wmma_f16(v16h a, v16h b, v8f c) {
  // (neg_a, A, neg_b, B, c_mod, C, reuse_a, reuse_b)
  return __builtin_amdgcn_wmma_f32_16x16x32_f16(false, a, false, b, (short)0, c, false, false);
}

// ---------------- fragment loaders (ISA 7.12.2 layouts) ----------------
// A 16x32: lane l -> row m = l&15, K in {klo..klo+7} u {klo+16..klo+23}, klo=8*(l>>4).
// B 32x16 from a TRANSPOSED [16 x 32] source has the identical access pattern
// (col n = l&15 instead of row): every fragment is two contiguous 16B loads/lane.

DEVINL v16h a_from_f16(const _Float16* base, int ld) {
  int l = lane(); int m = l & 15; int klo = (l >> 4) << 3;
  const _Float16* row = base + (size_t)m * ld;
  h8 lo = *(const h8*)(row + klo);
  h8 hi = *(const h8*)(row + klo + 16);
  v16h a;
#pragma unroll
  for (int e = 0; e < 8; ++e) { a[e] = lo[e]; a[8 + e] = hi[e]; }
  return a;
}

DEVINL v16h bT_from_f16(const _Float16* baseT, int ld) { return a_from_f16(baseT, ld); }

DEVINL v16h a_from_f16_pad16(const _Float16* base, int ld) {  // K<16 valid, rest 0
  int l = lane(); int m = l & 15; int klo = (l >> 4) << 3;
  h8 lo = *(const h8*)(base + (size_t)m * ld + klo);
  v16h a;
#pragma unroll
  for (int e = 0; e < 8; ++e) { a[e] = lo[e]; a[8 + e] = (_Float16)0.f; }
  return a;
}

DEVINL v16h a_from_f16_pad16_rows(const _Float16* base, int ld, int nrows) {
  int l = lane(); int m = l & 15; int klo = (l >> 4) << 3;
  v16h a;
  if (m < nrows) {
    h8 lo = *(const h8*)(base + (size_t)m * ld + klo);
#pragma unroll
    for (int e = 0; e < 8; ++e) a[e] = lo[e];
  } else {
#pragma unroll
    for (int e = 0; e < 8; ++e) a[e] = (_Float16)0.f;
  }
#pragma unroll
  for (int e = 0; e < 8; ++e) a[8 + e] = (_Float16)0.f;
  return a;
}

// B-frag from transposed [16 x 24] (spatial V): K rows >= 24 zeroed.
DEVINL v16h bT_from_f16_rows24(const _Float16* baseT, int ld) {
  int l = lane(); int n = l & 15; int klo = (l >> 4) << 3;
  const _Float16* row = baseT + (size_t)n * ld;
  h8 lo = *(const h8*)(row + klo);
  v16h b;
  if (klo == 0) {               // K {0..7,16..23} all valid
    h8 hi = *(const h8*)(row + 16);
#pragma unroll
    for (int e = 0; e < 8; ++e) { b[e] = lo[e]; b[8 + e] = hi[e]; }
  } else {                      // K {8..15} valid, {24..31} zero
#pragma unroll
    for (int e = 0; e < 8; ++e) { b[e] = lo[e]; b[8 + e] = (_Float16)0.f; }
  }
  return b;
}

DEVINL v16h a_from_f32_sum2(const float* p, const float* q, int ld) {
  int l = lane(); int m = l & 15; int klo = (l >> 4) << 3;
  const float* rp = p + (size_t)m * ld;
  const float* rq = q + (size_t)m * ld;
  f8v lo = *(const f8v*)(rp + klo)      + *(const f8v*)(rq + klo);
  f8v hi = *(const f8v*)(rp + klo + 16) + *(const f8v*)(rq + klo + 16);
  v16h a;
#pragma unroll
  for (int e = 0; e < 8; ++e) { a[e] = (_Float16)lo[e]; a[8 + e] = (_Float16)hi[e]; }
  return a;
}

// half-wave (16 lane) reductions: C-layout rows live inside one 16-lane half
DEVINL float hmax16(float v) {
#pragma unroll
  for (int m = 1; m < 16; m <<= 1) v = fmaxf(v, __shfl_xor(v, m, 32));
  return v;
}
DEVINL float hsum16(float v) {
#pragma unroll
  for (int m = 1; m < 16; m <<= 1) v += __shfl_xor(v, m, 32);
  return v;
}

// ========== kernel 0a: one-shot weight prep (f32 -> f16 transposed) ==========
// nb batched [R x C] row-major f32 -> nb batched [C x R] row-major f16
__global__ void __launch_bounds__(256) k_prep_w(const float* __restrict__ in,
                                                _Float16* __restrict__ out,
                                                int R, int C, int nb) {
  int total = nb * R * C;
  for (int i = blockIdx.x * 256 + (int)threadIdx.x; i < total; i += gridDim.x * 256) {
    int b = i / (R * C);
    int rc = i - b * (R * C);
    int r = rc / C, c = rc - r * C;
    out[(size_t)b * R * C + (size_t)c * R + r] = (_Float16)in[i];
  }
}

// ========== kernel 0b: one-shot x prep (f32 -> f16, same layout) ==========
__global__ void __launch_bounds__(256) k_prep_x(const float* __restrict__ in,
                                                _Float16* __restrict__ out, int total) {
  for (int i = blockIdx.x * 256 + (int)threadIdx.x; i < total; i += gridDim.x * 256)
    out[i] = (_Float16)in[i];
}

// ============ kernel 1: fused temporal+spatial QKV projections ============
// per (n,j): rows (b*S+s) of xh, [16x128]@[128x16] x6, one shared A fragment.
// temporal q,k row-major [B,N,J,S,D], v transposed [B,N,J,D,S];
// spatial  q,k row-major [B,S,N,J,D], v transposed [B,S,N,D,J].
__global__ void __launch_bounds__(32) k_qkv(
    const _Float16* __restrict__ xh,
    const _Float16* __restrict__ qwTt, const _Float16* __restrict__ kwTt,
    const _Float16* __restrict__ vwTt,  // [N*J, D, H]
    const _Float16* __restrict__ qwTs,  // [N*J, D, H]
    const _Float16* __restrict__ kwTs, const _Float16* __restrict__ vwTs,  // [N, D, H]
    _Float16* __restrict__ qoT, _Float16* __restrict__ koT, _Float16* __restrict__ vToT,
    _Float16* __restrict__ qoS, _Float16* __restrict__ koS, _Float16* __restrict__ vToS) {
  int tile = blockIdx.x;
  int bst = tile & 127;
  int j = (tile >> 7) % Jj;
  int n = tile / (128 * Jj);
  int r0 = bst * 16;
  const _Float16* ab = xh + ((size_t)r0 * Jj + j) * Hh;
  size_t wnj = (size_t)(n * Jj + j) * Hh * Dd;
  size_t wn  = (size_t)n * Hh * Dd;
  v8f cqt = {}, ckt = {}, cvt = {}, cqs = {}, cks = {}, cvs = {};
#pragma unroll
  for (int kk = 0; kk < Hh / 32; ++kk) {
    v16h a = a_from_f16(ab + kk * 32, Jj * Hh);
    cqt = wmma_f16(a, bT_from_f16(qwTt + wnj + kk * 32, Hh), cqt);
    ckt = wmma_f16(a, bT_from_f16(kwTt + wnj + kk * 32, Hh), ckt);
    cvt = wmma_f16(a, bT_from_f16(vwTt + wnj + kk * 32, Hh), cvt);
    cqs = wmma_f16(a, bT_from_f16(qwTs + wnj + kk * 32, Hh), cqs);
    cks = wmma_f16(a, bT_from_f16(kwTs + wn  + kk * 32, Hh), cks);
    cvs = wmma_f16(a, bT_from_f16(vwTs + wn  + kk * 32, Hh), cvs);
  }
  int l = lane(); int colc = l & 15; int half = l >> 4;
  int b = r0 / Ss; int s0 = r0 % Ss;
  size_t tq = ((size_t)(b * Nn + n) * Jj + j) * Ss * Dd + (size_t)s0 * Dd;
  size_t tv = ((size_t)(b * Nn + n) * Jj + j) * Dd * Ss;
#pragma unroll
  for (int r = 0; r < 8; ++r) {
    int m = r + 8 * half;
    qoT[tq + (size_t)m * Dd + colc] = (_Float16)cqt[r];
    koT[tq + (size_t)m * Dd + colc] = (_Float16)ckt[r];
    vToT[tv + (size_t)colc * Ss + (s0 + m)] = (_Float16)cvt[r];    // transposed
    size_t rid = (size_t)(r0 + m);
    size_t sq = ((rid * Nn + n) * Jj + j) * Dd;
    qoS[sq + colc] = (_Float16)cqs[r];
    koS[sq + colc] = (_Float16)cks[r];
    vToS[((rid * Nn + n) * Dd + colc) * Jj + j] = (_Float16)cvs[r]; // transposed
  }
}

// ============ kernel 2: temporal flash attention (causal) ============
__global__ void __launch_bounds__(32) k_attn_temporal(
    const _Float16* __restrict__ q, const _Float16* __restrict__ k,
    const _Float16* __restrict__ vT, _Float16* __restrict__ o_out /*[B,S,J,N*D]*/) {
  __shared__ __align__(16) _Float16 pbuf[16 * 32];
  int tile = blockIdx.x;
  int st = tile & 31;
  int j  = (tile >> 5) % Jj;
  int n  = (tile / (32 * Jj)) % Nn;
  int b  = tile / (32 * Jj * Nn);
  int s0 = st * 16;
  size_t slice = (size_t)(b * Nn + n) * Jj + j;
  const _Float16* qb  = q  + slice * Ss * Dd + (size_t)s0 * Dd;
  const _Float16* kb  = k  + slice * Ss * Dd;
  const _Float16* vTb = vT + slice * Dd * Ss;
  int l = lane(); int colc = l & 15; int half = l >> 4;

  v16h qa = a_from_f16_pad16(qb, Dd);          // [16 x (16 pad 32)]
  v8f o = {};
  float mrun[8], lrun[8];
#pragma unroll
  for (int r = 0; r < 8; ++r) { mrun[r] = NEGF; lrun[r] = 0.f; }

  const int tlim = s0 + 16;
  for (int tc = 0; tc < tlim; tc += 32) {      // causal upper bound
    if (tc + 32 < tlim) {                      // prefetch next K/V chunk
      __builtin_prefetch(kb + (size_t)(tc + 32) * Dd, 0, 1);
      __builtin_prefetch(vTb + tc + 32, 0, 1);
    }
    // score B-frags: K row-major [S x D] -> contiguous per-lane loads (K padded)
    v16h kb0 = a_from_f16_pad16(kb + (size_t)tc * Dd, Dd);
    v16h kb1 = a_from_f16_pad16(kb + (size_t)(tc + 16) * Dd, Dd);
    v8f sa = {}, sb = {};
    sa = wmma_f16(qa, kb0, sa);
    sb = wmma_f16(qa, kb1, sb);
#pragma unroll
    for (int r = 0; r < 8; ++r) {
      int srow = s0 + r + 8 * half;
      float x0 = (tc + colc      <= srow) ? sa[r] * SCALE : NEGF;
      float x1 = (tc + 16 + colc <= srow) ? sb[r] * SCALE : NEGF;
      float mnew = fmaxf(mrun[r], hmax16(fmaxf(x0, x1)));
      float alpha = __expf(mrun[r] - mnew);
      float p0 = __expf(x0 - mnew);
      float p1 = __expf(x1 - mnew);
      lrun[r] = lrun[r] * alpha + hsum16(p0 + p1);
      mrun[r] = mnew;
      o[r] = o[r] * alpha;
      int m = r + 8 * half;
      pbuf[m * 32 + colc]      = (_Float16)p0;
      pbuf[m * 32 + 16 + colc] = (_Float16)p1;
    }
    __syncthreads();                           // C-layout -> A-layout via LDS
    v16h pa = a_from_f16(pbuf, 32);
    v16h vf = bT_from_f16(vTb + tc, Ss);       // V transposed: contiguous loads
    o = wmma_f16(pa, vf, o);
    __syncthreads();
  }
#pragma unroll
  for (int r = 0; r < 8; ++r) {
    int m = r + 8 * half;
    float val = o[r] / lrun[r];
    o_out[((size_t)(b * Ss + s0 + m) * Jj + j) * Hh + n * Dd + colc] = (_Float16)val;
  }
}

// ================= kernel 3: spatial attention (J=24) =================
DEVINL void softmax16(v8f sa, v8f sb, _Float16* pb) {
  int l = lane(); int colc = l & 15; int half = l >> 4;
#pragma unroll
  for (int r = 0; r < 8; ++r) {
    float x0 = sa[r] * SCALE;
    float x1 = (colc < Jj - 16) ? sb[r] * SCALE : NEGF;   // mask cols >= 24
    float mx = hmax16(fmaxf(x0, x1));
    float p0 = __expf(x0 - mx);
    float p1 = __expf(x1 - mx);
    float inv = 1.f / hsum16(p0 + p1);
    int m = r + 8 * half;
    pb[m * 32 + colc]      = (_Float16)(p0 * inv);
    pb[m * 32 + 16 + colc] = (_Float16)(p1 * inv);
  }
}

__global__ void __launch_bounds__(32) k_attn_spatial(
    const _Float16* __restrict__ q, const _Float16* __restrict__ k,
    const _Float16* __restrict__ vT, _Float16* __restrict__ o_out /*[B,S,J,N*D]*/) {
  __shared__ __align__(16) _Float16 pbuf[32 * 32];
  int unit = blockIdx.x;
  int n = unit % Nn; size_t bs = (size_t)(unit / Nn);
  const _Float16* qb  = q  + (bs * Nn + n) * Jj * Dd;
  const _Float16* kb  = k  + (bs * Nn + n) * Jj * Dd;
  const _Float16* vTb = vT + (bs * Nn + n) * Dd * Jj;
  int l = lane(); int colc = l & 15; int half = l >> 4;

  v16h qa0 = a_from_f16_pad16(qb, Dd);
  v16h qa1 = a_from_f16_pad16_rows(qb + 16 * Dd, Dd, Jj - 16);
  v16h kb0 = a_from_f16_pad16(kb, Dd);                       // score B: rows of K
  v16h kb1 = a_from_f16_pad16_rows(kb + 16 * Dd, Dd, Jj - 16);
  v8f s00 = {}, s01 = {}, s10 = {}, s11 = {};
  s00 = wmma_f16(qa0, kb0, s00);
  s01 = wmma_f16(qa0, kb1, s01);
  s10 = wmma_f16(qa1, kb0, s10);
  s11 = wmma_f16(qa1, kb1, s11);
  softmax16(s00, s01, pbuf);
  softmax16(s10, s11, pbuf + 16 * 32);
  __syncthreads();
  v16h pa0 = a_from_f16(pbuf, 32);
  v16h pa1 = a_from_f16(pbuf + 16 * 32, 32);
  v16h vf  = bT_from_f16_rows24(vTb, Jj);      // V transposed [D x J], rows>=24 zeroed
  v8f o0 = {}, o1 = {};
  o0 = wmma_f16(pa0, vf, o0);
  o1 = wmma_f16(pa1, vf, o1);
#pragma unroll
  for (int r = 0; r < 8; ++r) {
    int m = r + 8 * half;
    o_out[(bs * Jj + m) * Hh + n * Dd + colc] = (_Float16)o0[r];
    int jr = 16 + m;
    if (jr < Jj)
      o_out[(bs * Jj + jr) * Hh + n * Dd + colc] = (_Float16)o1[r];
  }
}

// ====== kernel 4: head-merge projection + residual + layernorm ======
// temporal: WT per-joint (wstride=H*H, bias=null); spatial: shared WT (wstride=0, bias)
__global__ void __launch_bounds__(32) k_proj_ln(
    const _Float16* __restrict__ o_in /*[B,S,J,H]*/, const _Float16* __restrict__ WT,
    long wstride, const float* __restrict__ bias, const float* __restrict__ x,
    const float* __restrict__ lnw, const float* __restrict__ lnb,
    float* __restrict__ out) {
  int tile = blockIdx.x;
  int bst = tile & 127;
  int j = tile >> 7;
  int r0 = bst * 16;
  const int ld = Jj * Hh;
  const _Float16* ab = o_in + ((size_t)r0 * Jj + j) * Hh;
  const _Float16* wb = WT + (size_t)j * wstride;   // [Hout x Hin] f16
  const float* xb = x + ((size_t)r0 * Jj + j) * Hh;
  float* ob = out + ((size_t)r0 * Jj + j) * Hh;
  int l = lane(); int colc = l & 15; int half = l >> 4;

  v16h afr[4];
#pragma unroll
  for (int kk = 0; kk < 4; ++kk) afr[kk] = a_from_f16(ab + kk * 32, ld);
  v8f acc[8];
#pragma unroll
  for (int ct = 0; ct < 8; ++ct) {
    v8f c = {};
#pragma unroll
    for (int kk = 0; kk < 4; ++kk)
      c = wmma_f16(afr[kk], bT_from_f16(wb + (size_t)(ct * 16) * Hh + kk * 32, Hh), c);
    acc[ct] = c;
  }
#pragma unroll
  for (int r = 0; r < 8; ++r) {
    int m = r + 8 * half;
    float s = 0.f;
#pragma unroll
    for (int ct = 0; ct < 8; ++ct) {
      int cg = ct * 16 + colc;
      float val = acc[ct][r] + (bias ? bias[cg] : 0.f) + xb[(size_t)m * ld + cg];
      acc[ct][r] = val;
      s += val;
    }
    float mean = hsum16(s) * (1.f / Hh);
    float vs = 0.f;
#pragma unroll
    for (int ct = 0; ct < 8; ++ct) { float d = acc[ct][r] - mean; vs += d * d; }
    float inv = rsqrtf(hsum16(vs) * (1.f / Hh) + 1e-5f);
#pragma unroll
    for (int ct = 0; ct < 8; ++ct) {
      int cg = ct * 16 + colc;
      ob[(size_t)m * ld + cg] = (acc[ct][r] - mean) * inv * lnw[cg] + lnb[cg];
    }
  }
}

// ============ kernel 5: combine + FFN + final layernorm ============
__global__ void __launch_bounds__(32) k_ffn_ln(
    const float* __restrict__ t_out, const float* __restrict__ s_out,
    const _Float16* __restrict__ w1T /*[F x H]*/, const float* __restrict__ b1,
    const _Float16* __restrict__ w2T /*[H x F]*/, const float* __restrict__ b2,
    const float* __restrict__ lnw, const float* __restrict__ lnb,
    float* __restrict__ out) {
  __shared__ __align__(16) _Float16 h1s[16 * Ffn];
  int r0 = blockIdx.x * 16;
  const float* tb = t_out + (size_t)r0 * Hh;
  const float* sb = s_out + (size_t)r0 * Hh;
  int l = lane(); int colc = l & 15; int half = l >> 4;

  v16h afr[4];
#pragma unroll
  for (int kk = 0; kk < 4; ++kk)
    afr[kk] = a_from_f32_sum2(tb + kk * 32, sb + kk * 32, Hh);   // a = t_out + s_out
#pragma unroll
  for (int ct = 0; ct < Ffn / 16; ++ct) {
    v8f c = {};
#pragma unroll
    for (int kk = 0; kk < 4; ++kk)
      c = wmma_f16(afr[kk], bT_from_f16(w1T + (size_t)(ct * 16) * Hh + kk * 32, Hh), c);
#pragma unroll
    for (int r = 0; r < 8; ++r) {
      int m = r + 8 * half; int cg = ct * 16 + colc;
      h1s[m * Ffn + cg] = (_Float16)fmaxf(c[r] + b1[cg], 0.f);   // ReLU, stage for GEMM2
    }
  }
  __syncthreads();
  v8f zero = {};
  v8f acc[8];
#pragma unroll
  for (int ct = 0; ct < 8; ++ct) acc[ct] = zero;
#pragma unroll
  for (int kk = 0; kk < Ffn / 32; ++kk) {
    v16h pa = a_from_f16(h1s + kk * 32, Ffn);
#pragma unroll
    for (int ct = 0; ct < 8; ++ct)
      acc[ct] = wmma_f16(pa, bT_from_f16(w2T + (size_t)(ct * 16) * Ffn + kk * 32, Ffn), acc[ct]);
  }
#pragma unroll
  for (int r = 0; r < 8; ++r) {
    int m = r + 8 * half;
    float s = 0.f;
#pragma unroll
    for (int ct = 0; ct < 8; ++ct) {
      int cg = ct * 16 + colc;
      float a_res = tb[(size_t)m * Hh + cg] + sb[(size_t)m * Hh + cg];
      float val = acc[ct][r] + b2[cg] + a_res;
      acc[ct][r] = val;
      s += val;
    }
    float mean = hsum16(s) * (1.f / Hh);
    float vs = 0.f;
#pragma unroll
    for (int ct = 0; ct < 8; ++ct) { float d = acc[ct][r] - mean; vs += d * d; }
    float inv = rsqrtf(hsum16(vs) * (1.f / Hh) + 1e-5f);
#pragma unroll
    for (int ct = 0; ct < 8; ++ct) {
      int cg = ct * 16 + colc;
      out[(size_t)(r0 + m) * Hh + cg] = (acc[ct][r] - mean) * inv * lnw[cg] + lnb[cg];
    }
  }
}

// ============================= launcher =============================
extern "C" void kernel_launch(void* const* d_in, const int* in_sizes, int n_in,
                              void* d_out, int out_size, void* d_ws, size_t ws_size,
                              hipStream_t stream) {
  (void)in_sizes; (void)n_in; (void)out_size; (void)ws_size;
  const float* x        = (const float*)d_in[0];
  // d_in[1] = mask: causal mask is applied analytically in k_attn_temporal
  const float* q_t      = (const float*)d_in[2];
  const float* k_t      = (const float*)d_in[3];
  const float* v_t      = (const float*)d_in[4];
  const float* proj_t   = (const float*)d_in[5];
  const float* ln_t_w   = (const float*)d_in[6];
  const float* ln_t_b   = (const float*)d_in[7];
  const float* q_s      = (const float*)d_in[8];
  const float* k_s      = (const float*)d_in[9];
  const float* v_s      = (const float*)d_in[10];
  const float* proj_s_w = (const float*)d_in[11];
  const float* proj_s_b = (const float*)d_in[12];
  const float* ln_s_w   = (const float*)d_in[13];
  const float* ln_s_b   = (const float*)d_in[14];
  const float* ff1_w    = (const float*)d_in[15];
  const float* ff1_b    = (const float*)d_in[16];
  const float* ff2_w    = (const float*)d_in[17];
  const float* ff2_b    = (const float*)d_in[18];
  const float* ln_f_w   = (const float*)d_in[19];
  const float* ln_f_b   = (const float*)d_in[20];
  float* out = (float*)d_out;

  const size_t nE = (size_t)Bb * Ss * Jj * Hh;   // 6.29M elements (stage buffers)
  char* w = (char*)d_ws;
  auto take = [&](size_t bytes) { void* p = (void*)w; w += (bytes + 255) & ~(size_t)255; return p; };
  _Float16* xh  = (_Float16*)take(nE * 2);       // f16 copy of x for matmuls
  _Float16* qT  = (_Float16*)take(nE * 2);
  _Float16* kT  = (_Float16*)take(nE * 2);
  _Float16* vTt = (_Float16*)take(nE * 2);
  _Float16* oT  = (_Float16*)take(nE * 2);
  _Float16* qS  = (_Float16*)take(nE * 2);
  _Float16* kS  = (_Float16*)take(nE * 2);
  _Float16* vTs = (_Float16*)take(nE * 2);
  _Float16* oS  = (_Float16*)take(nE * 2);
  float* t_out  = (float*)take(nE * 4);
  float* s_out  = (float*)take(nE * 4);
  // transposed f16 weights
  const size_t nQW = (size_t)Nn * Jj * Hh * Dd;      // 393216
  _Float16* qwTt = (_Float16*)take(nQW * 2);
  _Float16* kwTt = (_Float16*)take(nQW * 2);
  _Float16* vwTt = (_Float16*)take(nQW * 2);
  _Float16* qwTs = (_Float16*)take(nQW * 2);
  _Float16* kwTs = (_Float16*)take((size_t)Nn * Hh * Dd * 2);
  _Float16* vwTs = (_Float16*)take((size_t)Nn * Hh * Dd * 2);
  _Float16* pwTt = (_Float16*)take((size_t)Jj * Hh * Hh * 2);
  _Float16* pwTs = (_Float16*)take((size_t)Hh * Hh * 2);
  _Float16* w1T  = (_Float16*)take((size_t)Hh * Ffn * 2);
  _Float16* w2T  = (_Float16*)take((size_t)Ffn * Hh * 2);

  dim3 blk(32);
  auto prep = [&](const float* in, _Float16* outp, int R, int C, int nb) {
    int total = nb * R * C;
    k_prep_w<<<dim3((total + 255) / 256), dim3(256), 0, stream>>>(in, outp, R, C, nb);
  };
  k_prep_x<<<dim3(4096), dim3(256), 0, stream>>>(x, xh, (int)nE);
  prep(q_t, qwTt, Hh, Dd, Nn * Jj);
  prep(k_t, kwTt, Hh, Dd, Nn * Jj);
  prep(v_t, vwTt, Hh, Dd, Nn * Jj);
  prep(q_s, qwTs, Hh, Dd, Nn * Jj);
  prep(k_s, kwTs, Hh, Dd, Nn);
  prep(v_s, vwTs, Hh, Dd, Nn);
  prep(proj_t, pwTt, Hh, Hh, Jj);
  prep(proj_s_w, pwTs, Hh, Hh, 1);
  prep(ff1_w, w1T, Hh, Ffn, 1);
  prep(ff2_w, w2T, Ffn, Hh, 1);

  k_qkv<<<dim3(Nn * Jj * 128), blk, 0, stream>>>(xh, qwTt, kwTt, vwTt, qwTs, kwTs, vwTs,
                                                 qT, kT, vTt, qS, kS, vTs);
  k_attn_temporal<<<dim3(Bb * Nn * Jj * (Ss / 16)), blk, 0, stream>>>(qT, kT, vTt, oT);
  k_attn_spatial <<<dim3(Bb * Ss * Nn), blk, 0, stream>>>(qS, kS, vTs, oS);
  k_proj_ln<<<dim3(Jj * 128), blk, 0, stream>>>(oT, pwTt, (long)Hh * Hh, nullptr,
                                                x, ln_t_w, ln_t_b, t_out);
  k_proj_ln<<<dim3(Jj * 128), blk, 0, stream>>>(oS, pwTs, 0L, proj_s_b,
                                                x, ln_s_w, ln_s_b, s_out);
  k_ffn_ln<<<dim3(Bb * Ss * Jj / 16), blk, 0, stream>>>(t_out, s_out, w1T, ff1_b,
                                                        w2T, ff2_b, ln_f_w, ln_f_b, out);
}
